// SimpleVideoTransformer_69956427317929
// MI455X (gfx1250) — matrix operand order repaired and checked by reading, not compiled, not run on hardware
//
#include <hip/hip_runtime.h>
#include <hip/hip_bf16.h>

typedef __attribute__((ext_vector_type(16))) __bf16 v16bf;
typedef __attribute__((ext_vector_type(8)))  float  v8f;

union FragAB { v16bf v; unsigned short u[16]; unsigned int d[8]; };
union FragC  { v8f v; float f[8]; };

__device__ __forceinline__ unsigned short f2bf(float x) {
    unsigned int u = __float_as_uint(x);
    u += 0x7fffu + ((u >> 16) & 1u);
    return (unsigned short)(u >> 16);
}
__device__ __forceinline__ float bf2f(unsigned short h) {
    return __uint_as_float(((unsigned int)h) << 16);
}

// ---- DPP lane reductions (VALU only, no LDS traffic) ----------------------
// quad_perm [1,0,3,2]=0xB1 (xor1), [2,3,0,1]=0x4E (xor2),
// ROW_HALF_MIRROR=0x141 (8-group), ROW_MIRROR=0x140 (16-group)
template<int C>
__device__ __forceinline__ float dpp_f(float x) {
    return __int_as_float(__builtin_amdgcn_mov_dpp(__float_as_int(x), C, 0xf, 0xf, true));
}
__device__ __forceinline__ float red16_max(float v) {
    v = fmaxf(v, dpp_f<0xB1>(v));
    v = fmaxf(v, dpp_f<0x4E>(v));
    v = fmaxf(v, dpp_f<0x141>(v));
    v = fmaxf(v, dpp_f<0x140>(v));
    return v;
}
__device__ __forceinline__ float red16_sum(float v) {
    v += dpp_f<0xB1>(v);
    v += dpp_f<0x4E>(v);
    v += dpp_f<0x141>(v);
    v += dpp_f<0x140>(v);
    return v;
}
__device__ __forceinline__ float red32_sum(float v) {
    v = red16_sum(v);
    v += __shfl_xor(v, 16, 32);          // single cross-half permute
    return v;
}

// ---------------------------------------------------------------------------
// Embedding: x = embed[tok] + actions@action_w + action_b + sinpos(t) + spatial_pos(n)
// ---------------------------------------------------------------------------
__global__ __launch_bounds__(256)
void embed_kernel(const int* __restrict__ tokens, const float* __restrict__ actions,
                  const float* __restrict__ embed, const float* __restrict__ action_w,
                  const float* __restrict__ action_b, const float* __restrict__ spatial_pos,
                  float* __restrict__ X)
{
    int idx = blockIdx.x * 256 + threadIdx.x;      // < 9216*512
    int d   = idx & 511;
    int tok = idx >> 9;
    int n   = tok % 576;
    int bt  = tok / 576;                            // b*8 + t
    int t   = bt & 7;
    int tv  = tokens[tok];
    float v = embed[(size_t)tv * 512 + d];
    const float* a = actions + (size_t)bt * 4;
    float ap = action_b[d];
#pragma unroll
    for (int i = 0; i < 4; i++) ap += a[i] * action_w[i * 512 + d];
    v += ap;
    {
        int dh = d & 255;
        float freq = __expf((float)dh * (-9.210340371976184f / 255.f));
        float e = (float)t * freq;
        v += (d < 256) ? __sinf(e) : __cosf(e);
    }
    v += spatial_pos[(size_t)n * 512 + d];
    X[idx] = v;
}

// ---------------------------------------------------------------------------
// LayerNorm over D=512, one wave per token, writes bf16
// ---------------------------------------------------------------------------
__global__ __launch_bounds__(256)
void ln_kernel(const float* __restrict__ X, const float* __restrict__ w,
               const float* __restrict__ bta, unsigned short* __restrict__ Y)
{
    int tokw = blockIdx.x * 8 + (threadIdx.x >> 5);
    int lane = threadIdx.x & 31;
    const float* x = X + (size_t)tokw * 512;
    float vals[16];
    float s = 0.f;
#pragma unroll
    for (int i = 0; i < 16; i++) { vals[i] = x[lane + i * 32]; s += vals[i]; }
    s = red32_sum(s);
    float mean = s * (1.f / 512.f);
    float vs = 0.f;
#pragma unroll
    for (int i = 0; i < 16; i++) { float dd = vals[i] - mean; vs += dd * dd; }
    vs = red32_sum(vs);
    float rstd = rsqrtf(vs * (1.f / 512.f) + 1e-5f);
    unsigned short* y = Y + (size_t)tokw * 512;
#pragma unroll
    for (int i = 0; i < 16; i++) {
        int d = lane + i * 32;
        y[d] = f2bf((vals[i] - mean) * rstd * w[d] + bta[d]);
    }
}

// ---------------------------------------------------------------------------
// Tiled WMMA GEMM: C[M,N] = A_bf16[M,K] @ W_f32[K,N] (+bias)(+resid)(gelu)
// block 256 thr = 8 waves, tile 128x128, K-step 32; waves in 4(M) x 2(N);
// each wave computes 32x64 via 2x4 wmma_f32_16x16x32_bf16 (8 wmma / K-step).
// MODE: 0 = bias -> bf16; 1 = bias + residual -> f32; 2 = bias+GELU -> bf16;
//       3 = (opt bias) -> f32.  WT: weight stored (N,K) row-major.
// ---------------------------------------------------------------------------
template<int MODE, bool WT>
__global__ __launch_bounds__(256)
void gemm_kernel(const unsigned short* __restrict__ A, int lda,
                 const float* __restrict__ W, int ldw,
                 const float* __restrict__ bias,
                 const float* __restrict__ resid,
                 void* __restrict__ out, int ldo,
                 int M, int N, int K)
{
    __shared__ unsigned short As[128][40];  // 80B rows: 16B-aligned
    __shared__ unsigned short Ws[128][36];  // stored transposed: [n][k]

    int tid  = threadIdx.x;
    int lane = tid & 31;
    int wave = tid >> 5;
    int wm = wave >> 1;          // 0..3
    int wn = wave & 1;           // 0..1
    int bm0 = blockIdx.x * 128;
    int bn0 = blockIdx.y * 128;
    int hi  = lane >> 4;
    int l15 = lane & 15;

    FragC acc[2][4];
#pragma unroll
    for (int i = 0; i < 2; i++)
#pragma unroll
        for (int j = 0; j < 4; j++)
#pragma unroll
            for (int r = 0; r < 8; r++) acc[i][j].f[r] = 0.f;

    // staging maps
    int arow = tid >> 1;             // 0..127
    int aks  = (tid & 1) * 16;       // 0 or 16
    int wnn  = tid & 127;            // 0..127 (coalesced over n)
    int wkq  = (tid >> 7) * 16;      // 0 or 16

    for (int k0 = 0; k0 < K; k0 += 32) {
        {   // stage A: 128x32 bf16, 16 halves per thread via 2x b128
            const unsigned short* src = A + (size_t)(bm0 + arow) * lda + k0 + aks;
            *(uint4*)&As[arow][aks]     = *(const uint4*)src;
            *(uint4*)&As[arow][aks + 8] = *(const uint4*)(src + 8);
        }
        {   // stage W transposed (coalesced over n for each k row)
            int gn = bn0 + wnn;
#pragma unroll
            for (int q = 0; q < 16; q++) {
                int kk = k0 + wkq + q;
                float wv = 0.f;
                if (gn < N) wv = WT ? W[(size_t)gn * ldw + kk] : W[(size_t)kk * ldw + gn];
                Ws[wnn][wkq + q] = f2bf(wv);
            }
        }
        __syncthreads();

        FragAB afr[2], bfr[4];
#pragma unroll
        for (int i = 0; i < 2; i++) {
            int row = wm * 32 + i * 16 + l15;
#pragma unroll
            for (int v = 0; v < 8; v++) {
                int kk = ((v >> 2) << 4) + hi * 8 + ((v & 3) << 1);
                afr[i].d[v] = *(const unsigned int*)&As[row][kk];
            }
        }
#pragma unroll
        for (int j = 0; j < 4; j++) {
            int col = wn * 64 + j * 16 + l15;
#pragma unroll
            for (int v = 0; v < 8; v++) {
                int kk = ((v >> 2) << 4) + hi * 8 + ((v & 3) << 1);
                bfr[j].d[v] = *(const unsigned int*)&Ws[col][kk];
            }
        }
#pragma unroll
        for (int i = 0; i < 2; i++)
#pragma unroll
            for (int j = 0; j < 4; j++)
                acc[i][j].v = __builtin_amdgcn_wmma_f32_16x16x32_bf16(
                    false, afr[i].v, false, bfr[j].v, (short)0, acc[i][j].v, false, false);
        __syncthreads();
    }

    // epilogue
#pragma unroll
    for (int i = 0; i < 2; i++) {
#pragma unroll
        for (int j = 0; j < 4; j++) {
            int col = bn0 + wn * 64 + j * 16 + l15;
            if (col >= N) continue;
            float bv = bias ? bias[col] : 0.f;
#pragma unroll
            for (int r = 0; r < 8; r++) {
                int row = bm0 + wm * 32 + i * 16 + r + hi * 8;
                float val = acc[i][j].f[r] + bv;
                if (MODE == 1) val += resid[(size_t)row * ldo + col];
                if (MODE == 2) val = 0.5f * val * (1.f + erff(val * 0.7071067811865475f));
                if (MODE == 0 || MODE == 2)
                    ((unsigned short*)out)[(size_t)row * ldo + col] = f2bf(val);
                else
                    ((float*)out)[(size_t)row * ldo + col] = val;
            }
        }
    }
}

// ---------------------------------------------------------------------------
// Spatial flash attention: 16 seqs x 576 tokens, 8 heads, HD=64.
// Block = 128 thr (4 waves), one wave per 16-query block; grid (9, 16, 8).
// K/V chunks (32 keys x 64 dims, bf16) staged into LDS with coalesced b128
// loads shared by the 4 waves; fragments gathered from LDS.
// ---------------------------------------------------------------------------
__global__ __launch_bounds__(128)
void attn_spatial_kernel(const unsigned short* __restrict__ QKV,
                         unsigned short* __restrict__ O)
{
    __shared__ unsigned short Ks[32][72];      // [key-in-chunk][dim], 144B rows
    __shared__ unsigned short Vs[32][72];      // [key-in-chunk][dim]
    __shared__ unsigned short P[4][16][36];    // per-wave probs

    int tid  = threadIdx.x;
    int lane = tid & 31;
    int wave = tid >> 5;                 // 0..3
    int hi = lane >> 4, l15 = lane & 15;
    int qb = blockIdx.x * 4 + wave;      // 0..35
    int m  = blockIdx.y;                 // 0..15
    int h  = blockIdx.z;                 // 0..7
    const int S = 576, C3 = 1536;
    size_t seqbase = (size_t)m * S;

    // Q fragments (per wave, loaded once from global)
    FragAB qa[2];
    {
        size_t tok = seqbase + qb * 16 + l15;
        const unsigned short* qp = QKV + tok * C3 + h * 64;
#pragma unroll
        for (int f = 0; f < 2; f++)
#pragma unroll
            for (int v = 0; v < 8; v++) {
                int kk = f * 32 + ((v >> 2) << 4) + hi * 8 + ((v & 3) << 1);
                qa[f].d[v] = *(const unsigned int*)(qp + kk);
            }
    }

    float mi[8], li[8];
    FragC accO[4];
#pragma unroll
    for (int r = 0; r < 8; r++) { mi[r] = -1e30f; li[r] = 0.f; }
#pragma unroll
    for (int t = 0; t < 4; t++)
#pragma unroll
        for (int r = 0; r < 8; r++) accO[t].f[r] = 0.f;

    int skey = tid >> 2;             // 0..31 : key within chunk
    int sds  = (tid & 3) * 16;       // 0,16,32,48 : dim start

    for (int c = 0; c < 18; c++) {               // 18 chunks of 32 keys
        __syncthreads();                          // protect Ks/Vs from overwrite
        {   // cooperative staging, coalesced b128
            const unsigned short* kp =
                QKV + (seqbase + c * 32 + skey) * C3 + 512 + h * 64 + sds;
            const unsigned short* vp = kp + 512;
            *(uint4*)&Ks[skey][sds]     = *(const uint4*)kp;
            *(uint4*)&Ks[skey][sds + 8] = *(const uint4*)(kp + 8);
            *(uint4*)&Vs[skey][sds]     = *(const uint4*)vp;
            *(uint4*)&Vs[skey][sds + 8] = *(const uint4*)(vp + 8);
        }
        __syncthreads();

        FragC sc[2];
#pragma unroll
        for (int half = 0; half < 2; half++) {
            FragAB kb[2];
#pragma unroll
            for (int f = 0; f < 2; f++)
#pragma unroll
                for (int v = 0; v < 8; v++) {
                    int kk = f * 32 + ((v >> 2) << 4) + hi * 8 + ((v & 3) << 1);      // dim pair (table order)
                    kb[f].d[v] = *(const unsigned int*)&Ks[half * 16 + l15][kk];
                }
            FragC z;
#pragma unroll
            for (int r = 0; r < 8; r++) z.f[r] = 0.f;
            z.v = __builtin_amdgcn_wmma_f32_16x16x32_bf16(false, qa[0].v, false, kb[0].v, (short)0, z.v, false, false);
            z.v = __builtin_amdgcn_wmma_f32_16x16x32_bf16(false, qa[1].v, false, kb[1].v, (short)0, z.v, false, false);
#pragma unroll
            for (int r = 0; r < 8; r++) sc[half].f[r] = z.f[r] * 0.125f;
        }
        // online softmax (row = r + 8*hi, cols spread over 16 lanes of group)
        float alpha[8];
#pragma unroll
        for (int r = 0; r < 8; r++) {
            float mv = red16_max(fmaxf(sc[0].f[r], sc[1].f[r]));
            float mnew = fmaxf(mi[r], mv);
            alpha[r] = __expf(mi[r] - mnew);
            float p0 = __expf(sc[0].f[r] - mnew);
            float p1 = __expf(sc[1].f[r] - mnew);
            sc[0].f[r] = p0; sc[1].f[r] = p1;
            float rs = red16_sum(p0 + p1);
            li[r] = li[r] * alpha[r] + rs;
            mi[r] = mnew;
        }
#pragma unroll
        for (int t = 0; t < 4; t++)
#pragma unroll
            for (int r = 0; r < 8; r++) accO[t].f[r] *= alpha[r];

        // re-layout P (C layout -> A layout) via per-wave LDS slice
#pragma unroll
        for (int half = 0; half < 2; half++)
#pragma unroll
            for (int r = 0; r < 8; r++)
                P[wave][r + hi * 8][half * 16 + l15] = f2bf(sc[half].f[r]);
        FragAB pa;
#pragma unroll
        for (int v = 0; v < 8; v++) {
            int kk = ((v >> 2) << 4) + hi * 8 + ((v & 3) << 1);
            pa.d[v] = *(const unsigned int*)&P[wave][l15][kk];
        }
        // P(16x32) @ V(32x64): 4 wmma, one per 16-dim output tile
#pragma unroll
        for (int t = 0; t < 4; t++) {
            FragAB vb;
#pragma unroll
            for (int v = 0; v < 8; v++) {
                int k1 = ((v >> 2) << 4) + hi * 8 + ((v & 3) << 1);               // key pair (rows of Vs)
                int dcol = t * 16 + l15;
                unsigned int lo = Vs[k1][dcol];
                unsigned int hb = Vs[k1 + 1][dcol];
                vb.d[v] = lo | (hb << 16);
            }
            accO[t].v = __builtin_amdgcn_wmma_f32_16x16x32_bf16(
                false, pa.v, false, vb.v, (short)0, accO[t].v, false, false);
        }
    }
#pragma unroll
    for (int t = 0; t < 4; t++)
#pragma unroll
        for (int r = 0; r < 8; r++) {
            size_t tok = seqbase + qb * 16 + r + hi * 8;
            O[tok * 512 + h * 64 + t * 16 + l15] = f2bf(accO[t].f[r] / li[r]);
        }
}

// ---------------------------------------------------------------------------
// Temporal attention: seqs (b,n) over T=8, causal. One wave per (b,n,h);
// each lane owns 2 head-dims; dots reduced across the wave.
// ---------------------------------------------------------------------------
__global__ __launch_bounds__(256)
void attn_temporal_kernel(const unsigned short* __restrict__ QKV,
                          unsigned short* __restrict__ O)
{
    int gid  = blockIdx.x * 8 + (threadIdx.x >> 5);   // wave id < 9216
    int lane = threadIdx.x & 31;
    int h = gid & 7;
    int n = (gid >> 3) % 576;
    int b = gid / (8 * 576);
    const int C3 = 1536;
    int d0 = lane * 2;

    float q0[8], q1[8], k0[8], k1[8], v0[8], v1[8];
#pragma unroll
    for (int t = 0; t < 8; t++) {
        size_t tok = ((size_t)(b * 8 + t)) * 576 + n;
        const unsigned short* base = QKV + tok * C3 + h * 64 + d0;
        q0[t] = bf2f(base[0]);    q1[t] = bf2f(base[1]);
        k0[t] = bf2f(base[512]);  k1[t] = bf2f(base[513]);
        v0[t] = bf2f(base[1024]); v1[t] = bf2f(base[1025]);
    }
#pragma unroll
    for (int tq = 0; tq < 8; tq++) {
        float s[8];
#pragma unroll
        for (int t = 0; t < 8; t++) {
            float d = red32_sum(q0[tq] * k0[t] + q1[tq] * k1[t]);
            s[t] = (t <= tq) ? d * 0.125f : -1e9f;
        }
        float mx = s[0];
#pragma unroll
        for (int t = 1; t < 8; t++) mx = fmaxf(mx, s[t]);
        float l = 0.f;
#pragma unroll
        for (int t = 0; t < 8; t++) { s[t] = __expf(s[t] - mx); l += s[t]; }
        float inv = 1.f / l;
        float o0 = 0.f, o1 = 0.f;
#pragma unroll
        for (int t = 0; t < 8; t++) { o0 += s[t] * v0[t]; o1 += s[t] * v1[t]; }
        size_t tok = ((size_t)(b * 8 + tq)) * 576 + n;
        unsigned short* op = O + tok * 512 + h * 64 + d0;
        op[0] = f2bf(o0 * inv);
        op[1] = f2bf(o1 * inv);
    }
}

// ---------------------------------------------------------------------------
extern "C" void kernel_launch(void* const* d_in, const int* in_sizes, int n_in,
                              void* d_out, int out_size, void* d_ws, size_t ws_size,
                              hipStream_t stream)
{
    const int*   tokens      = (const int*)d_in[0];
    const float* actions     = (const float*)d_in[1];
    // d_in[2] temporal_mask: causal mask implemented directly (same -1e9 numerics)
    const float* embed       = (const float*)d_in[3];
    const float* action_w    = (const float*)d_in[4];
    const float* action_b    = (const float*)d_in[5];
    const float* spatial_pos = (const float*)d_in[6];
    const float* ln_w        = (const float*)d_in[7];
    const float* ln_b        = (const float*)d_in[8];
    const float* qkv_w       = (const float*)d_in[9];
    const float* qkv_b       = (const float*)d_in[10];
    const float* proj_w      = (const float*)d_in[11];
    const float* proj_b      = (const float*)d_in[12];
    const float* mlp_w1      = (const float*)d_in[13];
    const float* mlp_b1      = (const float*)d_in[14];
    const float* mlp_w2      = (const float*)d_in[15];
    const float* mlp_b2      = (const float*)d_in[16];
    const float* out_ln_w    = (const float*)d_in[17];
    const float* out_ln_b    = (const float*)d_in[18];
    float* out = (float*)d_out;

    const int M = 9216;                     // B*T*N tokens
    char* ws = (char*)d_ws;
    float*          X   = (float*)ws;                                    // 9216*512 f32
    unsigned short* XN  = (unsigned short*)(ws + (size_t)M * 512 * 4);   // bf16 (LN out / attn out)
    unsigned short* QKV = (unsigned short*)((char*)XN + (size_t)M * 512 * 2);
    unsigned short* HID = (unsigned short*)((char*)QKV + (size_t)M * 1536 * 2);

    dim3 B256(256);

    embed_kernel<<<(M * 512) / 256, B256, 0, stream>>>(
        tokens, actions, embed, action_w, action_b, spatial_pos, X);

    for (int l = 0; l < 4; l++) {
        for (int s = 0; s < 2; s++) {
            ln_kernel<<<M / 8, B256, 0, stream>>>(
                X, ln_w + (size_t)(l * 3 + s) * 512, ln_b + (size_t)(l * 3 + s) * 512, XN);
            gemm_kernel<0, false><<<dim3(M / 128, 1536 / 128), B256, 0, stream>>>(
                XN, 512, qkv_w + (size_t)(l * 2 + s) * 512 * 1536, 1536,
                qkv_b + (size_t)(l * 2 + s) * 1536, nullptr, QKV, 1536, M, 1536, 512);
            if (s == 0)
                attn_spatial_kernel<<<dim3(9, 16, 8), dim3(128), 0, stream>>>(QKV, XN);
            else
                attn_temporal_kernel<<<M / 8, B256, 0, stream>>>(QKV, XN);
            gemm_kernel<1, false><<<dim3(M / 128, 512 / 128), B256, 0, stream>>>(
                XN, 512, proj_w + (size_t)(l * 2 + s) * 512 * 512, 512,
                proj_b + (size_t)(l * 2 + s) * 512, X, X, 512, M, 512, 512);
        }
        ln_kernel<<<M / 8, B256, 0, stream>>>(
            X, ln_w + (size_t)(l * 3 + 2) * 512, ln_b + (size_t)(l * 3 + 2) * 512, XN);
        gemm_kernel<2, false><<<dim3(M / 128, 2048 / 128), B256, 0, stream>>>(
            XN, 512, mlp_w1 + (size_t)l * 512 * 2048, 2048,
            mlp_b1 + (size_t)l * 2048, nullptr, HID, 2048, M, 2048, 512);
        gemm_kernel<1, false><<<dim3(M / 128, 512 / 128), B256, 0, stream>>>(
            HID, 2048, mlp_w2 + (size_t)l * 2048 * 512, 512,
            mlp_b2 + (size_t)l * 512, X, X, 512, M, 512, 2048);
    }

    ln_kernel<<<M / 8, B256, 0, stream>>>(X, out_ln_w, out_ln_b, XN);
    // logits = XN @ embed^T  (N=17, weight transposed)
    gemm_kernel<3, true><<<dim3(M / 128, 1), B256, 0, stream>>>(
        XN, 512, embed, 512, nullptr, nullptr, out, 17, M, 17, 512);
}
